// e_dist_20890720927800
// MI455X (gfx1250) — compile-verified, matrix operand order checked by reading, not run on hardware
//
#include <hip/hip_runtime.h>

// Problem constants (match reference)
#define WAY 64
#define SHOT 16
#define SEG 8
#define DIM 2048
#define NQ 4096
#define NS (WAY * SHOT)  // 1024 supports

// Phase-2 tiling
#define TM 128      // query rows per workgroup (8 waves x 16)
#define TN 128      // support cols per workgroup (8 classes x 16)
#define KB 32       // K per LDS stage (one bf16 WMMA K-step)
#define BSTRIDE 40  // padded per-col K stride in LDS elems (80B: 16B-aligned,
                    // 20-word bank stride -> 16 lanes hit 16 distinct banks)

typedef float v8f  __attribute__((ext_vector_type(8)));
typedef __bf16 v16bf __attribute__((ext_vector_type(16)));
typedef __bf16 v8bf  __attribute__((ext_vector_type(8)));

// ---------------------------------------------------------------------------
// Phase 1: segment mean over SEG=8, split into bf16 hi/lo planes
// (x = hi + lo with |lo| <= 2^-9 |x|), plus exact f32 squared norm.
// One workgroup per row; fully coalesced reads/writes.
// ---------------------------------------------------------------------------
__global__ void __launch_bounds__(256)
seg_mean_split_kernel(const float* __restrict__ in, __bf16* __restrict__ hi,
                      __bf16* __restrict__ lo, float* __restrict__ sq) {
  const int row = blockIdx.x;
  const int tid = threadIdx.x;
  const float* base = in + (size_t)row * SEG * DIM;

  float local_sq = 0.0f;
#pragma unroll
  for (int j = 0; j < DIM / 256; ++j) {
    const int d = tid + j * 256;
    float acc = 0.0f;
#pragma unroll
    for (int s = 0; s < SEG; ++s) acc += base[(size_t)s * DIM + d];
    const float m = acc * (1.0f / (float)SEG);
    const __bf16 h = (__bf16)m;
    const __bf16 l = (__bf16)(m - (float)h);
    hi[(size_t)row * DIM + d] = h;
    lo[(size_t)row * DIM + d] = l;
    local_sq += m * m;  // norms stay exact f32
  }

  __shared__ float red[256];
  red[tid] = local_sq;
  __syncthreads();
  for (int off = 128; off > 0; off >>= 1) {
    if (tid < off) red[tid] += red[tid + off];
    __syncthreads();
  }
  if (tid == 0) sq[row] = red[0];
}

// ---------------------------------------------------------------------------
// Phase 2: split-bf16 WMMA GEMM (cross = hi·hi + hi·lo + lo·hi, f32 accum)
// fused with sqrt + per-class mean epilogue.
//
// Workgroup: 256 threads (8 waves). Tile: 128 rows x 128 cols (8 classes).
// B hi/lo tiles are double-buffered in LDS and shared by all 8 waves;
// A fragments are wave-private and loaded straight from global (L2-resident).
//
// bf16 A/B fragment layout per V_WMMA_F32_16X16X32_BF16 (wave32):
//   lane L -> row/col (L&15); K chunks [half*8,+8) and [16+half*8,+8)
//   where half = L>>4  ->  two 16B loads per lane per matrix per K-step.
// C/D: VGPR i -> M = i (lanes 0-15) / i+8 (lanes 16-31), N = L&15.
// ---------------------------------------------------------------------------
__global__ void __launch_bounds__(256)
dist_gemm_bf16x3_kernel(const __bf16* __restrict__ qhi, const __bf16* __restrict__ qlo,
                        const __bf16* __restrict__ shi, const __bf16* __restrict__ slo,
                        const float* __restrict__ qsq, const float* __restrict__ ssq,
                        float* __restrict__ out) {
  __shared__ __bf16 Bs[2][2][TN][BSTRIDE];  // [buf][plane][col][k] = 40 KB

  const int tid  = threadIdx.x;
  const int lane = tid & 31;
  const int wave = tid >> 5;
  const int half = lane >> 4;  // 0: lanes 0-15, 1: lanes 16-31
  const int l16  = lane & 15;

  const int row0 = blockIdx.y * TM + wave * 16;  // this wave's query rows
  const int col0 = blockIdx.x * TN;              // first support col of block

  const __bf16* ahi = qhi + (size_t)(row0 + l16) * DIM;
  const __bf16* alo = qlo + (size_t)(row0 + l16) * DIM;
  const int kA0 = half * 8;       // first 8-elem K chunk for this lane
  const int kA1 = 16 + half * 8;  // second 8-elem K chunk

  v8f acc[8];
#pragma unroll
  for (int j = 0; j < 8; ++j) acc[j] = v8f{};

  // Stage B hi/lo tiles [col0, col0+TN) x [kt, kt+KB) into LDS buffer `buf`.
  // 2 planes * 128 cols * 4 chunks = 1024 b128 chunks; 4 per thread.
  auto stage = [&](int buf, int kt) {
#pragma unroll
    for (int t = 0; t < 4; ++t) {
      const int idx   = tid + t * 256;  // 0..1023
      const int plane = idx >> 9;       // 0 = hi, 1 = lo
      const int rem   = idx & 511;
      const int col   = rem >> 2;       // 0..127
      const int kc    = rem & 3;        // 8-elem chunk within KB
      const __bf16* src =
          (plane ? slo : shi) + (size_t)(col0 + col) * DIM + kt + kc * 8;
      const v8bf v = *(const v8bf*)src;
      *(v8bf*)&Bs[buf][plane][col][kc * 8] = v;
    }
  };

  stage(0, 0);
  __syncthreads();

  int s = 0;
  for (int kt = 0; kt < DIM; kt += KB, s ^= 1) {
    if (kt + KB < DIM) stage(s ^ 1, kt + KB);  // overlap next-stage fill

    // Wave-private A hi/lo fragments for this K-step (global, L2-resident).
    const v8bf ah0 = *(const v8bf*)(ahi + kt + kA0);
    const v8bf ah1 = *(const v8bf*)(ahi + kt + kA1);
    const v8bf al0 = *(const v8bf*)(alo + kt + kA0);
    const v8bf al1 = *(const v8bf*)(alo + kt + kA1);
    const v16bf Ah = __builtin_shufflevector(ah0, ah1, 0, 1, 2, 3, 4, 5, 6, 7,
                                             8, 9, 10, 11, 12, 13, 14, 15);
    const v16bf Al = __builtin_shufflevector(al0, al1, 0, 1, 2, 3, 4, 5, 6, 7,
                                             8, 9, 10, 11, 12, 13, 14, 15);

#pragma unroll
    for (int j = 0; j < 8; ++j) {
      const int c = j * 16 + l16;
      const v8bf bh0 = *(const v8bf*)&Bs[s][0][c][kA0];
      const v8bf bh1 = *(const v8bf*)&Bs[s][0][c][kA1];
      const v8bf bl0 = *(const v8bf*)&Bs[s][1][c][kA0];
      const v8bf bl1 = *(const v8bf*)&Bs[s][1][c][kA1];
      const v16bf Bh = __builtin_shufflevector(bh0, bh1, 0, 1, 2, 3, 4, 5, 6, 7,
                                               8, 9, 10, 11, 12, 13, 14, 15);
      const v16bf Bl = __builtin_shufflevector(bl0, bl1, 0, 1, 2, 3, 4, 5, 6, 7,
                                               8, 9, 10, 11, 12, 13, 14, 15);
      // cross += Ah*Bh + Ah*Bl + Al*Bh   (lo*lo dropped: ~2^-18 rel)
      acc[j] = __builtin_amdgcn_wmma_f32_16x16x32_bf16(false, Ah, false, Bh,
                                                       (short)0, acc[j], false, false);
      acc[j] = __builtin_amdgcn_wmma_f32_16x16x32_bf16(false, Ah, false, Bl,
                                                       (short)0, acc[j], false, false);
      acc[j] = __builtin_amdgcn_wmma_f32_16x16x32_bf16(false, Al, false, Bh,
                                                       (short)0, acc[j], false, false);
    }
    __syncthreads();  // reads of Bs[s] done; stores into Bs[s^1] visible
  }

  // Epilogue: dist = sqrt(max(q^2 + s^2 - 2*cross, 1e-12)); mean over the
  // 16 supports of each class (the 16 N-lanes), negate, write [Q, WAY].
  float qs[8];
#pragma unroll
  for (int i = 0; i < 8; ++i) qs[i] = qsq[row0 + i + 8 * half];

#pragma unroll
  for (int j = 0; j < 8; ++j) {
    const float ss = ssq[col0 + j * 16 + l16];
#pragma unroll
    for (int i = 0; i < 8; ++i) {
      float d2 = qs[i] + ss - 2.0f * acc[j][i];
      d2 = fmaxf(d2, 1e-12f);
      float dist = sqrtf(d2);
      // xor masks < 16 stay inside each 16-lane half -> per-row class sums
      dist += __shfl_xor(dist, 1, 32);
      dist += __shfl_xor(dist, 2, 32);
      dist += __shfl_xor(dist, 4, 32);
      dist += __shfl_xor(dist, 8, 32);
      if (l16 == 0) {
        const int row = row0 + i + 8 * half;
        out[(size_t)row * WAY + (blockIdx.x * 8 + j)] = -dist * (1.0f / (float)SHOT);
      }
    }
  }
}

// ---------------------------------------------------------------------------
// Launch. Workspace layout:
//   s_hi [NS*DIM bf16] | s_lo [NS*DIM] | q_hi [NQ*DIM] | q_lo [NQ*DIM]
//   | s_sq [NS f32] | q_sq [NQ f32]                       (~42 MB)
// ---------------------------------------------------------------------------
extern "C" void kernel_launch(void* const* d_in, const int* in_sizes, int n_in,
                              void* d_out, int out_size, void* d_ws, size_t ws_size,
                              hipStream_t stream) {
  (void)in_sizes; (void)n_in; (void)out_size; (void)ws_size;
  const float* support = (const float*)d_in[0];
  // d_in[1] = support_labels (balanced: repeat(arange(64),16) by construction)
  const float* queries = (const float*)d_in[2];
  // d_in[3] = way (compile-time WAY here)

  __bf16* s_hi = (__bf16*)d_ws;
  __bf16* s_lo = s_hi + (size_t)NS * DIM;
  __bf16* q_hi = s_lo + (size_t)NS * DIM;
  __bf16* q_lo = q_hi + (size_t)NQ * DIM;
  float*  s_sq = (float*)(q_lo + (size_t)NQ * DIM);
  float*  q_sq = s_sq + NS;

  seg_mean_split_kernel<<<NS, 256, 0, stream>>>(support, s_hi, s_lo, s_sq);
  seg_mean_split_kernel<<<NQ, 256, 0, stream>>>(queries, q_hi, q_lo, q_sq);

  dim3 grid(WAY / 8, NQ / TM);  // 8 class-groups x 32 row-blocks
  dist_gemm_bf16x3_kernel<<<grid, 256, 0, stream>>>(q_hi, q_lo, s_hi, s_lo,
                                                    q_sq, s_sq, (float*)d_out);
}
